// Head_21878563406324
// MI455X (gfx1250) — matrix operand order repaired and checked by reading, not compiled, not run on hardware
//
#include <hip/hip_runtime.h>
#include <hip/hip_bf16.h>

typedef __attribute__((ext_vector_type(16))) _Float16 v16h;
typedef __attribute__((ext_vector_type(8)))  float    v8f;
typedef unsigned int u32x4 __attribute__((ext_vector_type(4)));
typedef int          i32x8 __attribute__((ext_vector_type(8)));
typedef int          i32x4 __attribute__((ext_vector_type(4)));

#define TDIM 4096
#define CDIM 1024
#define HDIM 64
#define BDIM 4

#if defined(__has_builtin)
#if __has_builtin(__builtin_amdgcn_tensor_load_to_lds)
#define USE_TDM 1
#endif
#endif

// ---- CDNA5 async global->LDS copy (ASYNCcnt) — fallback staging path ----
__device__ __forceinline__ void async_ld_b128(const _Float16* lds_dst, const _Float16* gsrc) {
    unsigned int       l = (unsigned int)(uintptr_t)lds_dst;   // LDS addr = low 32b of generic ptr
    unsigned long long g = (unsigned long long)(uintptr_t)gsrc;
    asm volatile("global_load_async_to_lds_b128 %0, %1, off" :: "v"(l), "v"(g) : "memory");
}
__device__ __forceinline__ void async_wait0() {
    asm volatile("s_wait_asynccnt 0x0" ::: "memory");
}
__device__ __forceinline__ void stage64_async(_Float16* dst, const _Float16* src,
                                              size_t rstride, int tid) {
#pragma unroll
    for (int it = 0; it < 4; ++it) {
        int ch = tid + it * 128;
        int r = ch >> 3, c8 = (ch & 7) << 3;
        async_ld_b128(&dst[r * 72 + c8], src + (size_t)r * rstride + c8);
    }
}

#ifdef USE_TDM
// ---- Tensor Data Mover: one instruction copies a 64x64 f16 tile into LDS,
// hardware-padding each 128B row by 16B -> 144B (72-half) LDS stride (ISA §8).
// clang-23 toolchain: 6-arg builtin (g0, g1, g2, g3, g4, cpol).
__device__ __forceinline__ void tdm_load_tile64(const _Float16* lds_dst,
                                                const _Float16* gsrc,
                                                unsigned int rstride /*elements*/) {
    unsigned int       lds = (unsigned int)(uintptr_t)lds_dst;
    unsigned long long ga  = (unsigned long long)(uintptr_t)gsrc;
    u32x4 g0;
    g0[0] = 1u;                                       // count=1 (valid user D#)
    g0[1] = lds;                                      // lds_addr (bytes)
    g0[2] = (unsigned int)ga;                         // global_addr[31:0]
    g0[3] = (unsigned int)(ga >> 32) | (2u << 30);    // global_addr[56:32] | type=2
    i32x8 g1;
    g1[0] = (1 << 16)                                 // data_size = 2 bytes
          | (1 << 20)                                 // pad_enable
          | (4 << 22)                                 // pad_interval: 32 DWORDs (=row)
          | (3 << 25);                                // pad_amount: 4 DWORDs (=16B)
    g1[1] = (int)((rstride & 0xFFFFu) << 16);         // tensor_dim0[15:0]
    g1[2] = (int)((rstride >> 16) | 0xFFFF0000u);     // tensor_dim0[31:16] | tensor_dim1[15:0]
    g1[3] = (64 << 16);                               // tensor_dim1[31:16]=0 | tile_dim0=64
    g1[4] = 64;                                       // tile_dim1=64, tile_dim2=0
    g1[5] = (int)rstride;                             // tensor_dim0_stride[31:0]
    g1[6] = 0;                                        // stride[47:32] | dim1_stride[15:0]
    g1[7] = 0;
    i32x4 gz4 = {0, 0, 0, 0};                         // groups 2/3 unused (2D tile)
    i32x8 gz8 = {0, 0, 0, 0, 0, 0, 0, 0};            // extra group (clang-23 form) unused
    __builtin_amdgcn_tensor_load_to_lds(g0, g1, gz4, gz4, gz8, 0);
}
#endif

// ---- WMMA fragment builders (contiguous k = k0 + 16*(lane/16) + s in both A and B, wave32) ----
__device__ __forceinline__ v16h load_afrag(const _Float16* src, int ld, int m0, int k0, int lane) {
    const int hi = lane >> 4, ln = lane & 15;
    const _Float16* row = src + (size_t)(m0 + ln) * ld;
    const int kb = k0 + hi * 16;
    v16h a;
#pragma unroll
    for (int j = 0; j < 8; ++j) {
        int kk = kb + 2 * j;
        a[2 * j]     = row[kk];
        a[2 * j + 1] = row[kk + 1];
    }
    return a;
}
__device__ __forceinline__ v16h load_bfrag(const _Float16* srcT, int ld, int n0, int k0, int lane) {
    const int hi = lane >> 4, ln = lane & 15;
    const _Float16* col = srcT + (size_t)(n0 + ln) * ld;
    const int kb = k0 + hi * 16;
    v16h b;
#pragma unroll
    for (int j = 0; j < 8; ++j) {
        b[2 * j]     = col[kb + 2 * j];
        b[2 * j + 1] = col[kb + 2 * j + 1];
    }
    return b;
}

#define WMMA_F16(A, B, C) \
    __builtin_amdgcn_wmma_f32_16x16x32_f16(false, (A), false, (B), (short)0, (C), false, false)

// ---------------- Kernel 1: fused QKV projection ----------------
// grid (BT/64, 3), block 128. y==0: q (scaled by 1/32), y==1: k, y==2: v (transposed out).
// Global->register prefetch double-buffers the K-loop staging behind the WMMAs.
__global__ __launch_bounds__(128) void qkv_proj(const float* __restrict__ x,
                                                const float* __restrict__ Wq,
                                                const float* __restrict__ Wk,
                                                const float* __restrict__ Wv,
                                                _Float16* __restrict__ qf,
                                                _Float16* __restrict__ kf,
                                                _Float16* __restrict__ vfT) {
    __shared__ _Float16 xs[64 * 36];
    __shared__ _Float16 wt[64 * 36];
    __shared__ _Float16 tr[64 * 72];

    const int tid = threadIdx.x;
    const int lane = tid & 31, wv = tid >> 5;
    const int hi = lane >> 4, ln = lane & 15;
    const int r0 = blockIdx.x * 64;

    const float* W = (blockIdx.y == 0) ? Wq : (blockIdx.y == 1) ? Wk : Wv;

    v8f acc[4] = {};
    float4 xr[4], wr[4];

    auto load_tiles = [&](int k0) {
#pragma unroll
        for (int it = 0; it < 4; ++it) {
            int ch = tid + it * 128;
            int r = ch >> 3, c4 = (ch & 7) << 2;
            xr[it] = *(const float4*)&x[(size_t)(r0 + r) * CDIM + k0 + c4];
            int k = ch >> 4, n4 = (ch & 15) << 2;
            wr[it] = *(const float4*)&W[(size_t)(k0 + k) * HDIM + n4];
        }
    };

    load_tiles(0);
    for (int k0 = 0; k0 < CDIM; k0 += 32) {
        __syncthreads();
#pragma unroll
        for (int it = 0; it < 4; ++it) {
            int ch = tid + it * 128;
            int r = ch >> 3, c4 = (ch & 7) << 2;
            _Float16* d = &xs[r * 36 + c4];
            d[0] = (_Float16)xr[it].x; d[1] = (_Float16)xr[it].y;
            d[2] = (_Float16)xr[it].z; d[3] = (_Float16)xr[it].w;
            int k = ch >> 4, n4 = (ch & 15) << 2;
            wt[(n4 + 0) * 36 + k] = (_Float16)wr[it].x;
            wt[(n4 + 1) * 36 + k] = (_Float16)wr[it].y;
            wt[(n4 + 2) * 36 + k] = (_Float16)wr[it].z;
            wt[(n4 + 3) * 36 + k] = (_Float16)wr[it].w;
        }
        __syncthreads();
        if (k0 + 32 < CDIM) load_tiles(k0 + 32);   // prefetch next step

        v16h a = load_afrag(xs, 36, wv * 16, 0, lane);
#pragma unroll
        for (int nt = 0; nt < 4; ++nt) {
            v16h b = load_bfrag(wt, 36, nt * 16, 0, lane);
            acc[nt] = WMMA_F16(a, b, acc[nt]);
        }
    }

    if (blockIdx.y == 2) {
        __syncthreads();
#pragma unroll
        for (int nt = 0; nt < 4; ++nt)
#pragma unroll
            for (int r = 0; r < 8; ++r)
                tr[(nt * 16 + ln) * 72 + wv * 16 + r + hi * 8] = (_Float16)acc[nt][r];
        __syncthreads();
        const int b = r0 / TDIM, t0 = r0 % TDIM;
        _Float16* vb = vfT + (size_t)b * HDIM * TDIM + t0;
#pragma unroll
        for (int it = 0; it < 4; ++it) {
            int ch = tid + it * 128;
            int h = ch >> 3, t8 = (ch & 7) << 3;
            *(uint4*)&vb[(size_t)h * TDIM + t8] = *(const uint4*)&tr[h * 72 + t8];
        }
    } else {
        const float osc = (blockIdx.y == 0) ? 0.03125f : 1.0f;   // fold C^-0.5 into q
        _Float16* out = (blockIdx.y == 0) ? qf : kf;
#pragma unroll
        for (int nt = 0; nt < 4; ++nt)
#pragma unroll
            for (int r = 0; r < 8; ++r) {
                int row = r0 + wv * 16 + r + hi * 8;
                out[(size_t)row * HDIM + nt * 16 + ln] = (_Float16)(acc[nt][r] * osc);
            }
    }
}

// ---------------- Kernel 2: causal flash attention ----------------
// grid (T/64, B), block 128 (4 waves, 16 query rows each).
// k/v tiles double-buffered in LDS; staged by the TDM (TENSORcnt) when available,
// otherwise by async global->LDS b128 copies (ASYNCcnt). Overlapped with WMMAs.
__global__ __launch_bounds__(128) void flash_attn(const _Float16* __restrict__ qf,
                                                  const _Float16* __restrict__ kf,
                                                  const _Float16* __restrict__ vfT,
                                                  float* __restrict__ out) {
    __shared__ _Float16 qs[64 * 72];
    __shared__ _Float16 kt[2][64 * 72];
    __shared__ _Float16 vt[2][64 * 72];
    __shared__ _Float16 pl[4][16 * 72];

    const int tid = threadIdx.x;
    const int lane = tid & 31, wv = tid >> 5;
    const int hi = lane >> 4, ln = lane & 15;
    const int b = blockIdx.y;
    const int i0 = blockIdx.x * 64;

    const _Float16* qb  = qf  + (size_t)b * TDIM * HDIM;
    const _Float16* kb  = kf  + (size_t)b * TDIM * HDIM;
    const _Float16* vTb = vfT + (size_t)b * HDIM * TDIM;
    float* ob = out + (size_t)b * TDIM * HDIM;

    // prologue: stage q tile + first k/v tiles
#ifdef USE_TDM
    if (wv == 0) {
        tdm_load_tile64(qs,    qb + (size_t)i0 * HDIM, HDIM);
        tdm_load_tile64(kt[0], kb,                     HDIM);
        tdm_load_tile64(vt[0], vTb,                    TDIM);
        __builtin_amdgcn_s_wait_tensorcnt(0);
    }
#else
    stage64_async(qs,    qb + (size_t)i0 * HDIM, HDIM, tid);
    stage64_async(kt[0], kb,                     HDIM, tid);
    stage64_async(vt[0], vTb,                    TDIM, tid);
    async_wait0();
#endif
    __syncthreads();

    const v16h qa0 = load_afrag(qs, 72, wv * 16, 0, lane);
    const v16h qa1 = load_afrag(qs, 72, wv * 16, 32, lane);

    v8f acc[4] = {};
    float mrow[8], lrow[8];
#pragma unroll
    for (int r = 0; r < 8; ++r) { mrow[r] = -1e30f; lrow[r] = 0.f; }

    const int nT = i0 / 64 + 1;                 // causal: tiles up to the diagonal
    for (int jt = 0; jt < nT; ++jt) {
        const int buf = jt & 1;
        if (jt + 1 < nT) {                      // prefetch next k/v tile
            const int j1 = (jt + 1) * 64;
#ifdef USE_TDM
            if (wv == 0) {
                tdm_load_tile64(kt[buf ^ 1], kb + (size_t)j1 * HDIM, HDIM);
                tdm_load_tile64(vt[buf ^ 1], vTb + j1,               TDIM);
            }
#else
            stage64_async(kt[buf ^ 1], kb + (size_t)j1 * HDIM, HDIM, tid);
            stage64_async(vt[buf ^ 1], vTb + j1,               TDIM, tid);
#endif
        }
        const _Float16* ktb = kt[buf];
        const _Float16* vtb = vt[buf];
        const int j0 = jt * 64;

        // S = (q/32) @ k^T  (16 x 64 keys, fp32 accum)
        v8f s[4];
#pragma unroll
        for (int nt = 0; nt < 4; ++nt) {
            v8f c = {};
            c = WMMA_F16(qa0, load_bfrag(ktb, 72, nt * 16, 0, lane), c);
            c = WMMA_F16(qa1, load_bfrag(ktb, 72, nt * 16, 32, lane), c);
            s[nt] = c;
        }

        // causal mask (finite -inf: no NaN)
#pragma unroll
        for (int nt = 0; nt < 4; ++nt) {
            int key = j0 + nt * 16 + ln;
#pragma unroll
            for (int r = 0; r < 8; ++r) {
                int qrow = i0 + wv * 16 + r + hi * 8;
                if (key > qrow) s[nt][r] = -1e30f;
            }
        }

        // online softmax: each row lives in the 16 lanes of a half-wave
#pragma unroll
        for (int r = 0; r < 8; ++r) {
            float mx = fmaxf(fmaxf(s[0][r], s[1][r]), fmaxf(s[2][r], s[3][r]));
#pragma unroll
            for (int off = 8; off >= 1; off >>= 1)
                mx = fmaxf(mx, __shfl_xor(mx, off, 16));
            float mn = fmaxf(mrow[r], mx);
            float alpha = __expf(mrow[r] - mn);
            mrow[r] = mn;
            float psum = 0.f;
#pragma unroll
            for (int nt = 0; nt < 4; ++nt) {
                float p = __expf(s[nt][r] - mn);
                s[nt][r] = p;
                psum += p;
            }
#pragma unroll
            for (int off = 8; off >= 1; off >>= 1)
                psum += __shfl_xor(psum, off, 16);
            lrow[r] = lrow[r] * alpha + psum;
#pragma unroll
            for (int nt = 0; nt < 4; ++nt) acc[nt][r] *= alpha;
        }

        // P: C-layout -> A-layout via LDS
        _Float16* pw = &pl[wv][0];
#pragma unroll
        for (int nt = 0; nt < 4; ++nt)
#pragma unroll
            for (int r = 0; r < 8; ++r)
                pw[(r + hi * 8) * 72 + nt * 16 + ln] = (_Float16)s[nt][r];
        __syncthreads();

        v16h pa0 = load_afrag(pw, 72, 0, 0, lane);
        v16h pa1 = load_afrag(pw, 72, 0, 32, lane);
#pragma unroll
        for (int nt = 0; nt < 4; ++nt) {
            acc[nt] = WMMA_F16(pa0, load_bfrag(vtb, 72, nt * 16, 0, lane), acc[nt]);
            acc[nt] = WMMA_F16(pa1, load_bfrag(vtb, 72, nt * 16, 32, lane), acc[nt]);
        }

        if (jt + 1 < nT) {                      // prefetched tile landed
#ifdef USE_TDM
            if (wv == 0) __builtin_amdgcn_s_wait_tensorcnt(0);
#else
            async_wait0();
#endif
        }
        __syncthreads();
    }

    // epilogue: normalize, fp32 store
#pragma unroll
    for (int nt = 0; nt < 4; ++nt)
#pragma unroll
        for (int r = 0; r < 8; ++r) {
            int row = i0 + wv * 16 + r + hi * 8;
            ob[(size_t)row * HDIM + nt * 16 + ln] = acc[nt][r] / lrow[r];
        }
}

extern "C" void kernel_launch(void* const* d_in, const int* in_sizes, int n_in,
                              void* d_out, int out_size, void* d_ws, size_t ws_size,
                              hipStream_t stream) {
    const float* x  = (const float*)d_in[0];
    const float* Wk = (const float*)d_in[1];
    const float* Wq = (const float*)d_in[2];
    const float* Wv = (const float*)d_in[3];
    float* out = (float*)d_out;

    const size_t per = (size_t)BDIM * TDIM * HDIM;   // 2 MB each (f16)
    _Float16* qf  = (_Float16*)d_ws;
    _Float16* kf  = qf + per;
    _Float16* vfT = kf + per;                         // [B][H][T] transposed

    dim3 blk(128);
    dim3 g1(BDIM * TDIM / 64, 3);
    qkv_proj<<<g1, blk, 0, stream>>>(x, Wq, Wk, Wv, qf, kf, vfT);

    dim3 g2(TDIM / 64, BDIM);
    flash_attn<<<g2, blk, 0, stream>>>(qf, kf, vfT, out);
}
